// PostPointNet_66778151518368
// MI455X (gfx1250) — matrix operand order repaired and checked by reading, not compiled, not run on hardware
//
#include <hip/hip_runtime.h>
#include <stdint.h>

// ---------------------------------------------------------------------------
// PostPointNet for MI455X (gfx1250, wave32).
// Pipeline: stable counting sort by ROI -> per-ROI FPS (LDS-resident coords)
// -> feature gather -> 5x [GEMM (v_wmma_f32_16x16x32_bf16) + BatchNorm(+ReLU)].
// Linear biases are dropped: BN subtracts the column mean, so a constant
// per-column bias cancels exactly (variance also unchanged).
// ---------------------------------------------------------------------------

#define NUM_POINTS 50
#define PMAX 4096
#define FPT (PMAX / 256)   // dist slots per thread
#define KPAD 4512          // 4500 padded to multiple of 32
#define EPSF 1e-5f

typedef __attribute__((ext_vector_type(16))) __bf16 v16bf;
typedef __attribute__((ext_vector_type(8)))  float  v8f;

// ---- fp32 -> bf16 (round to nearest even), packed pair into one dword -----
__device__ __forceinline__ unsigned int f2bf_pack(float lo, float hi) {
  union { float f; unsigned int u; } a, b;
  a.f = lo; b.f = hi;
  unsigned int ul = (a.u + 0x7FFFu + ((a.u >> 16) & 1u)) >> 16;
  unsigned int uh = (b.u + 0x7FFFu + ((b.u >> 16) & 1u)) >> 16;
  return (uh << 16) | (ul & 0xFFFFu);
}

// ---------------------------------------------------------------------------
// 1) Per-block ROI histogram (256 points per block, 256 bins).
// ---------------------------------------------------------------------------
__global__ __launch_bounds__(256) void k_hist(const int* __restrict__ roi, int N,
                                              int* __restrict__ blockHist) {
  __shared__ int bins[256];
  int tid = threadIdx.x;
  bins[tid] = 0;
  __syncthreads();
  int p = blockIdx.x * 256 + tid;
  if (p < N) atomicAdd(&bins[roi[p]], 1);
  __syncthreads();
  blockHist[blockIdx.x * 256 + tid] = bins[tid];
}

// ---------------------------------------------------------------------------
// 2) Scan: per-ROI prefix over blocks + cross-ROI exclusive scan.
//    blockHist becomes the absolute base offset per (block, roi).
// ---------------------------------------------------------------------------
__global__ __launch_bounds__(256) void k_scan(int* __restrict__ blockHist, int B,
                                              int* __restrict__ counts,
                                              int* __restrict__ offsets) {
  __shared__ int tot[256];
  int r = threadIdx.x;
  int run = 0;
  for (int b = 0; b < B; ++b) {
    int t = blockHist[b * 256 + r];
    blockHist[b * 256 + r] = run;
    run += t;
  }
  tot[r] = run;
  __syncthreads();
  int offv = 0;
  for (int i = 0; i < r; ++i) offv += tot[i];
  counts[r] = run;
  offsets[r] = offv;
  for (int b = 0; b < B; ++b) blockHist[b * 256 + r] += offv;
}

// ---------------------------------------------------------------------------
// 3) Stable scatter: intra-block rank = #earlier points in block w/ same ROI.
//    Reproduces jnp.argsort(idx) (stable) grouping.
// ---------------------------------------------------------------------------
__global__ __launch_bounds__(256) void k_scatter(const int* __restrict__ roi, int N,
                                                 const int* __restrict__ blockBase,
                                                 int* __restrict__ order) {
  __shared__ int sroi[256];
  int tid = threadIdx.x;
  int p = blockIdx.x * 256 + tid;
  int myroi = (p < N) ? roi[p] : -1;
  sroi[tid] = myroi;
  __syncthreads();
  if (p < N) {
    int rank = 0;
    for (int j = 0; j < tid; ++j) rank += (sroi[j] == myroi) ? 1 : 0;
    order[blockBase[blockIdx.x * 256 + myroi] + rank] = p;
  }
}

// ---------------------------------------------------------------------------
// 4) FPS + feature gather. One workgroup (8 waves) per ROI. Coords in LDS,
//    running min-dist in registers. First-max tie-break (== jnp.argmax).
//    Writes featbuf[R][KPAD] fp32, zero-padded (and zero rows for empty ROIs).
// ---------------------------------------------------------------------------
__global__ __launch_bounds__(256) void k_fps(const int* __restrict__ counts,
                                             const int* __restrict__ offsets,
                                             const int* __restrict__ order,
                                             const float* __restrict__ coords,
                                             const float* __restrict__ feats,
                                             float* __restrict__ featbuf) {
  __shared__ float sx[PMAX], sy[PMAX], sz[PMAX];
  __shared__ float redV[256];
  __shared__ int   redI[256];
  __shared__ int   samp[NUM_POINTS];
  __shared__ int   s_last;
  int r = blockIdx.x, tid = threadIdx.x;
  int cntFull = counts[r];
  int off = offsets[r];
  if (cntFull <= 0) {  // counts>0 mask in reference -> zero row
    for (int t = tid; t < KPAD; t += 256) featbuf[(size_t)r * KPAD + t] = 0.0f;
    return;
  }
  int cnt = cntFull < PMAX ? cntFull : PMAX;

  float dl[FPT];
  for (int i = 0; i < FPT; ++i) {
    int s = tid + i * 256;
    if (s < cnt) {
      int g = order[off + s];
      sx[s] = coords[g * 3 + 0];
      sy[s] = coords[g * 3 + 1];
      sz[s] = coords[g * 3 + 2];
      dl[i] = 1e10f;
    } else {
      sx[s] = 0.0f; sy[s] = 0.0f; sz[s] = 0.0f;
      dl[i] = -1.0f;  // invalid slots pinned to -1, never selected
    }
  }
  if (tid == 0) samp[0] = 0;
  __syncthreads();

  int last = 0;
  for (int k = 1; k < NUM_POINTS; ++k) {
    float lx = sx[last], ly = sy[last], lz = sz[last];
    float bv = -2.0f; int bi = 0;
    for (int i = 0; i < FPT; ++i) {
      int s = tid + i * 256;
      float dx = sx[s] - lx, dy = sy[s] - ly, dz = sz[s] - lz;
      float d = dx * dx + dy * dy + dz * dz;
      float nd = (s < cnt) ? fminf(dl[i], d) : -1.0f;
      dl[i] = nd;
      if (nd > bv) { bv = nd; bi = s; }  // ascending s: strict > keeps first max
    }
    redV[tid] = bv; redI[tid] = bi;
    __syncthreads();
    for (int stride = 128; stride > 0; stride >>= 1) {
      if (tid < stride) {
        float v2 = redV[tid + stride]; int i2 = redI[tid + stride];
        if (v2 > redV[tid] || (v2 == redV[tid] && i2 < redI[tid])) {
          redV[tid] = v2; redI[tid] = i2;
        }
      }
      __syncthreads();
    }
    if (tid == 0) { s_last = redI[0]; samp[k] = s_last; }
    __syncthreads();
    last = s_last;
  }

  // gather [NUM_POINTS x 90] features, zero-pad to KPAD
  for (int t = tid; t < KPAD; t += 256) {
    float v = 0.0f;
    if (t < NUM_POINTS * 90) {
      int j = t / 90, f = t - j * 90;
      int g = order[off + samp[j]];
      v = feats[g * 90 + f];
    }
    featbuf[(size_t)r * KPAD + t] = v;
  }
}

// ---------------------------------------------------------------------------
// 5) Pack fp32 activations into WMMA A-fragment layout (bf16).
//    Blob (mt, ks): 32 lanes x 8 dwords, contiguous 32B per lane.
//    A 16-bit 16x32 layout: lane<16 holds K in [0..7]U[16..23] of row lane&15;
//    lane>=16 holds K in [8..15]U[24..31]. (ISA 7.12.2)
// ---------------------------------------------------------------------------
__global__ __launch_bounds__(256) void k_pack_a(const float* __restrict__ x,
                                                unsigned int* __restrict__ pa,
                                                int Cols, int nK) {
  int idx = blockIdx.x * 256 + threadIdx.x;
  int reg = idx & 7;
  int lane = (idx >> 3) & 31;
  int lin = idx >> 8;
  int ks = lin % nK;
  int mt = lin / nK;
  int m = mt * 16 + (lane & 15);
  int k = ks * 32 + (reg & 3) * 2 + ((lane >> 4) & 1) * 8 + (reg >> 2) * 16;
  float lo = (k     < Cols) ? x[(size_t)m * Cols + k]     : 0.0f;
  float hi = (k + 1 < Cols) ? x[(size_t)m * Cols + k + 1] : 0.0f;
  pa[idx] = f2bf_pack(lo, hi);
}

// ---------------------------------------------------------------------------
// 6) Pack fp32 weights [K x N] into WMMA B-fragment layout (bf16).
//    B 16-bit 32x16: lanes 0-15 hold K in [0..7]U[16..23], lanes 16-31
//    K in [8..15]U[24..31], consecutive K pair per dword.
// ---------------------------------------------------------------------------
__global__ __launch_bounds__(256) void k_pack_b(const float* __restrict__ w,
                                                unsigned int* __restrict__ pb,
                                                int K, int N, int nK) {
  int idx = blockIdx.x * 256 + threadIdx.x;
  int reg = idx & 7;
  int lane = (idx >> 3) & 31;
  int lin = idx >> 8;
  int ks = lin % nK;
  int nt = lin / nK;
  int n = nt * 16 + (lane & 15);
  int k = ks * 32 + (reg & 3) * 2 + ((lane >> 4) & 1) * 8 + (reg >> 2) * 16;
  float lo = (k     < K) ? w[(size_t)k * N + n]       : 0.0f;
  float hi = (k + 1 < K) ? w[(size_t)(k + 1) * N + n] : 0.0f;
  pb[idx] = f2bf_pack(lo, hi);
}

// ---------------------------------------------------------------------------
// 7) GEMM: one wave per 16x16 output tile, K-loop of v_wmma_f32_16x16x32_bf16.
//    Fragments are contiguous 32B/lane -> two global_load_b128 per operand.
// ---------------------------------------------------------------------------
__global__ __launch_bounds__(32) void k_gemm(const unsigned int* __restrict__ pa,
                                             const unsigned int* __restrict__ pb,
                                             float* __restrict__ out,
                                             int nK, int N) {
  int mt = blockIdx.x, nt = blockIdx.y, lane = threadIdx.x;
  const v16bf* A = reinterpret_cast<const v16bf*>(pa) + (size_t)mt * nK * 32 + lane;
  const v16bf* B = reinterpret_cast<const v16bf*>(pb) + (size_t)nt * nK * 32 + lane;
  v8f acc = {0.f, 0.f, 0.f, 0.f, 0.f, 0.f, 0.f, 0.f};
  for (int k = 0; k < nK; ++k) {
    v16bf a = A[(size_t)k * 32];
    v16bf b = B[(size_t)k * 32];
    acc = __builtin_amdgcn_wmma_f32_16x16x32_bf16(
        /*neg_a=*/false, a, /*neg_b=*/false, b,
        /*c_mod=*/(short)0, acc, /*reuse_a=*/false, /*reuse_b=*/false);
  }
  // D layout: lane l, vgpr v -> m = 8*(l>=16)+v, n = l&15 (ISA 7.12.2)
  int n = nt * 16 + (lane & 15);
  int mbase = mt * 16 + ((lane >> 4) & 1) * 8;
  for (int v = 0; v < 8; ++v) out[(size_t)(mbase + v) * N + n] = acc[v];
}

// ---------------------------------------------------------------------------
// 8) BatchNorm (training stats over R rows, biased var) + optional ReLU.
//    One thread per output column.
// ---------------------------------------------------------------------------
__global__ __launch_bounds__(256) void k_bn(const float* __restrict__ x,
                                            float* __restrict__ y,
                                            const float* __restrict__ g,
                                            const float* __restrict__ be,
                                            int Ncols, int R, int relu) {
  int c = blockIdx.x * 256 + threadIdx.x;
  if (c >= Ncols) return;
  float sum = 0.0f, sq = 0.0f;
  for (int r = 0; r < R; ++r) {
    float v = x[(size_t)r * Ncols + c];
    sum += v; sq += v * v;
  }
  float invR = 1.0f / (float)R;
  float m = sum * invR;
  float var = sq * invR - m * m;
  float s = g[c] * rsqrtf(var + EPSF);
  float bb = be[c];
  for (int r = 0; r < R; ++r) {
    float v = (x[(size_t)r * Ncols + c] - m) * s + bb;
    if (relu) v = fmaxf(v, 0.0f);
    y[(size_t)r * Ncols + c] = v;
  }
}

// ---------------------------------------------------------------------------
extern "C" void kernel_launch(void* const* d_in, const int* in_sizes, int n_in,
                              void* d_out, int out_size, void* d_ws, size_t ws_size,
                              hipStream_t stream) {
  (void)n_in; (void)ws_size;
  const int*   roiIdx = (const int*)  d_in[1];
  const float* feats  = (const float*)d_in[2];
  const float* coords = (const float*)d_in[3];
  const float* w1 = (const float*)d_in[4];
  const float* g1 = (const float*)d_in[6],  *be1 = (const float*)d_in[7];
  const float* w2 = (const float*)d_in[8];
  const float* g2 = (const float*)d_in[10], *be2 = (const float*)d_in[11];
  const float* w3 = (const float*)d_in[12];
  const float* g3 = (const float*)d_in[14], *be3 = (const float*)d_in[15];
  const float* w4 = (const float*)d_in[16];
  const float* g4 = (const float*)d_in[18], *be4 = (const float*)d_in[19];
  const float* w5 = (const float*)d_in[20];
  const float* g5 = (const float*)d_in[22], *be5 = (const float*)d_in[23];
  float* outp = (float*)d_out;

  int N = in_sizes[1];           // 500000 points
  int R = out_size / 512;        // 256 ROIs (output is [R,512])
  int B = (N + 255) / 256;       // sort blocks
  int MT = R / 16;               // M tiles for GEMM (16)
  const int nK1 = KPAD / 32;     // 141 K-steps for layer 1

  // workspace carve-out (256B aligned)
  char* ws = (char*)d_ws;
  size_t woff = 0;
  auto take = [&](size_t bytes) -> char* {
    char* p = ws + woff;
    woff = (woff + bytes + 255) & ~(size_t)255;
    return p;
  };
  int*          blockHist = (int*)         take((size_t)B * 256 * 4);
  int*          counts    = (int*)         take(256 * 4);
  int*          offsets   = (int*)         take(256 * 4);
  int*          order     = (int*)         take((size_t)N * 4);
  float*        featbuf   = (float*)       take((size_t)R * KPAD * 4);
  unsigned int* pa        = (unsigned int*)take((size_t)MT * nK1 * 1024);
  unsigned int* pb        = (unsigned int*)take((size_t)16 * nK1 * 1024);
  float*        zbuf      = (float*)       take((size_t)R * 512 * 4);
  float*        hbuf      = (float*)       take((size_t)R * 512 * 4);

  // ---- stable grouping + FPS + gather ----
  k_hist   <<<B, 256, 0, stream>>>(roiIdx, N, blockHist);
  k_scan   <<<1, 256, 0, stream>>>(blockHist, B, counts, offsets);
  k_scatter<<<B, 256, 0, stream>>>(roiIdx, N, blockHist, order);
  k_fps    <<<R, 256, 0, stream>>>(counts, offsets, order, coords, feats, featbuf);

  // ---- L1: [R,4500] x [4500,256], BN+ReLU ----
  k_pack_a<<<MT * nK1, 256, 0, stream>>>(featbuf, pa, KPAD, nK1);
  k_pack_b<<<16 * nK1, 256, 0, stream>>>(w1, pb, 4500, 256, nK1);
  k_gemm<<<dim3(MT, 16), 32, 0, stream>>>(pa, pb, zbuf, nK1, 256);
  k_bn<<<1, 256, 0, stream>>>(zbuf, hbuf, g1, be1, 256, R, 1);

  // ---- L2: [R,256] x [256,256], BN+ReLU ----
  k_pack_a<<<MT * 8, 256, 0, stream>>>(hbuf, pa, 256, 8);
  k_pack_b<<<16 * 8, 256, 0, stream>>>(w2, pb, 256, 256, 8);
  k_gemm<<<dim3(MT, 16), 32, 0, stream>>>(pa, pb, zbuf, 8, 256);
  k_bn<<<1, 256, 0, stream>>>(zbuf, hbuf, g2, be2, 256, R, 1);

  // ---- L3: [R,256] x [256,512], BN (no ReLU) ----
  k_pack_a<<<MT * 8, 256, 0, stream>>>(hbuf, pa, 256, 8);
  k_pack_b<<<32 * 8, 256, 0, stream>>>(w3, pb, 256, 512, 8);
  k_gemm<<<dim3(MT, 32), 32, 0, stream>>>(pa, pb, zbuf, 8, 512);
  k_bn<<<2, 256, 0, stream>>>(zbuf, hbuf, g3, be3, 512, R, 0);

  // ---- L4: [R,512] x [512,256], BN+ReLU ----
  k_pack_a<<<MT * 16, 256, 0, stream>>>(hbuf, pa, 512, 16);
  k_pack_b<<<16 * 16, 256, 0, stream>>>(w4, pb, 512, 256, 16);
  k_gemm<<<dim3(MT, 16), 32, 0, stream>>>(pa, pb, zbuf, 16, 256);
  k_bn<<<1, 256, 0, stream>>>(zbuf, hbuf, g4, be4, 256, R, 1);

  // ---- L5: [R,256] x [256,512], BN+ReLU -> d_out ----
  k_pack_a<<<MT * 8, 256, 0, stream>>>(hbuf, pa, 256, 8);
  k_pack_b<<<32 * 8, 256, 0, stream>>>(w5, pb, 256, 512, 8);
  k_gemm<<<dim3(MT, 32), 32, 0, stream>>>(pa, pb, zbuf, 8, 512);
  k_bn<<<2, 256, 0, stream>>>(zbuf, outp, g5, be5, 512, R, 1);
}
